// STOMPnet2_16355235463735
// MI455X (gfx1250) — matrix operand order repaired and checked by reading, not compiled, not run on hardware
//
#include <hip/hip_runtime.h>
#include <hip/hip_bf16.h>

// ---------------------------------------------------------------------------
// STOMPnet routed MoE MLP for MI455X (gfx1250), bf16 WMMA path.
//
// B=128, G=128, E=8, S=512, DG=64, H=1024, A=16, DIN=576.
// Round 2: M-tile = 64 rows per workgroup -> each W fragment feeds 4 WMMAs
// (4 row-tiles per wave), cutting L2 weight traffic 4x vs M=16 and raising
// the wmma:vmem ratio in the inner loop. A tiles staged in the 320KB WGP LDS.
// ---------------------------------------------------------------------------

typedef __attribute__((ext_vector_type(16))) __bf16 v16bf;
typedef __attribute__((ext_vector_type(8)))  __bf16 v8bf;
typedef __attribute__((ext_vector_type(8)))  float  v8f;

#define B_    128
#define G_    128
#define E_    8
#define S_    512
#define DG_   64
#define H_    1024
#define A_    16
#define DIN_  576
#define NPAIR (B_ * G_)   // 16384
#define MT    64          // rows per workgroup in layer1/layer2

// ---------------- workspace layout (bytes) ----------------
static constexpr size_t OFF_CNT  = 0;                              // 8 ints
static constexpr size_t OFF_LIST = 256;                            // E*NPAIR ints
static constexpr size_t SZ_LIST  = (size_t)E_ * NPAIR * 4;         // 512 KB
static constexpr size_t OFF_W1T  = OFF_LIST + SZ_LIST;
static constexpr size_t SZ_W1T   = (size_t)E_ * H_ * DIN_ * 2;     // 9.4 MB  [E][N=H][K=DIN] bf16
static constexpr size_t OFF_W2T  = OFF_W1T + SZ_W1T;
static constexpr size_t SZ_W2T   = (size_t)E_ * H_ * H_ * 2;       // 16.8 MB [E][N=H][K=H]
static constexpr size_t OFF_W3T  = OFF_W2T + SZ_W2T;
static constexpr size_t SZ_W3T   = (size_t)E_ * A_ * H_ * 2;       // 256 KB  [E][N=A][K=H]
static constexpr size_t OFF_H1   = OFF_W3T + SZ_W3T;
static constexpr size_t SZ_H     = (size_t)NPAIR * H_ * 2;         // 33.5 MB bf16
static constexpr size_t OFF_H2   = OFF_H1 + SZ_H;                  // total ~94 MB

// ---------------- WMMA helper ----------------
__device__ __forceinline__ v8f wmma_bf16(v16bf a, v16bf b, v8f c) {
  // D(16x16 f32) = A(16x32 bf16) * B(32x16 bf16) + C
  return __builtin_amdgcn_wmma_f32_16x16x32_bf16(
      /*neg_a=*/false, a, /*neg_b=*/false, b,
      /*c_mod=*/(short)0, c, /*reuse_a=*/false, /*reuse_b=*/false);
}

// Build A fragment (16x32 bf16) from a row-major bf16 row pointer.
// Layout: lane m = lane&15; half = lane>>4; elements 0..15 = K[half*16 .. half*16+15]
// (contiguous, matching the B fragment's k order).
__device__ __forceinline__ v16bf load_a_frag(const __bf16* row, int kb, int half) {
  v8bf a0 = *(const v8bf*)(row + kb + half * 16);
  v8bf a1 = *(const v8bf*)(row + kb + half * 16 + 8);
  return __builtin_shufflevector(a0, a1, 0,1,2,3,4,5,6,7,8,9,10,11,12,13,14,15);
}

// ---------------- kernel 0: zero expert counters ----------------
__global__ void init_k(int* __restrict__ cnt) {
  if (threadIdx.x < E_) cnt[threadIdx.x] = 0;
}

// ---------------- kernel 1: gumbel-argmax routing + list build ----------------
__global__ __launch_bounds__(256) void route_k(const float* __restrict__ logits,  // [G,E]
                                               const float* __restrict__ u,       // [B,G,E]
                                               int* __restrict__ cnt,
                                               int* __restrict__ lists) {
  int p = blockIdx.x * 256 + threadIdx.x;       // pair id = b*G + g, 0..16383
  int g = p & (G_ - 1);
  const float eps = 1e-10f;
  float best = -__builtin_inff();
  int be = 0;
  #pragma unroll
  for (int e = 0; e < E_; ++e) {
    float uu  = u[(size_t)p * E_ + e];
    float gum = -__logf(-__logf(uu + eps) + eps);
    float s   = logits[g * E_ + e] + gum;
    if (s > best) { best = s; be = e; }         // strict >: first-max, matches jnp.argmax
  }
  int pos = atomicAdd(&cnt[be], 1);
  lists[be * NPAIR + pos] = p;
}

// ---------------- kernel 2: fp32 [E][K][N] -> bf16 transposed [E][N][K] ----------------
// grid = (N/16, K/16, E), block = 256. LDS tile transpose, coalesced both sides.
__global__ __launch_bounds__(256) void transpose_bf16_k(const float* __restrict__ W,
                                                        __bf16* __restrict__ WT,
                                                        int K, int N) {
  __shared__ float tile[16][17];
  int e  = blockIdx.z;
  int n0 = blockIdx.x * 16, k0 = blockIdx.y * 16;
  int tn = threadIdx.x & 15, tk = threadIdx.x >> 4;
  const float* We  = W  + (size_t)e * K * N;
  __bf16*      WTe = WT + (size_t)e * N * K;
  tile[tk][tn] = We[(size_t)(k0 + tk) * N + (n0 + tn)];
  __syncthreads();
  WTe[(size_t)(n0 + tk) * K + (k0 + tn)] = (__bf16)tile[tn][tk];
}

// ---------------- shared GEMM epilogue/compute for layer1/layer2 ----------------
// Block = 256 threads (8 waves), tile = MT(64) rows x N cols.
// Each wave owns 16-col N-tiles (8 passes) and 4 row-tiles per N-tile, so each
// B fragment (32x16 of W^T) is reused by 4 WMMAs.
template<int K>
__device__ __forceinline__ void gemm_tile_compute(const __bf16* __restrict__ WT_e, // [N][K]
                                                  const float* __restrict__ bias_e,
                                                  const __bf16 (*At)[K],           // [MT][K] LDS
                                                  const int* rows,                 // [MT] LDS
                                                  __bf16* __restrict__ Hout,       // [NPAIR][N]
                                                  int N, int t) {
  int wave = t >> 5, lane = t & 31;
  int m = lane & 15, half = lane >> 4;
  for (int pass = 0; pass < N / 16 / 8; ++pass) {
    int nt = wave + pass * 8;
    v8f acc0 = {0.f,0.f,0.f,0.f,0.f,0.f,0.f,0.f};
    v8f acc1 = {0.f,0.f,0.f,0.f,0.f,0.f,0.f,0.f};
    v8f acc2 = {0.f,0.f,0.f,0.f,0.f,0.f,0.f,0.f};
    v8f acc3 = {0.f,0.f,0.f,0.f,0.f,0.f,0.f,0.f};
    // B fragment: lane = column (nt*16 + m), 16 contiguous K values at half*16.
    const __bf16* wb = WT_e + (size_t)(nt * 16 + m) * K + half * 16;
    for (int kb = 0; kb < K; kb += 32) {
      v16bf bv = *(const v16bf*)(wb + kb);
      acc0 = wmma_bf16(load_a_frag(&At[ 0 + m][0], kb, half), bv, acc0);
      acc1 = wmma_bf16(load_a_frag(&At[16 + m][0], kb, half), bv, acc1);
      acc2 = wmma_bf16(load_a_frag(&At[32 + m][0], kb, half), bv, acc2);
      acc3 = wmma_bf16(load_a_frag(&At[48 + m][0], kb, half), bv, acc3);
    }
    int n = nt * 16 + m;
    float bias = bias_e[n];
    v8f accs[4] = {acc0, acc1, acc2, acc3};
    #pragma unroll
    for (int r4 = 0; r4 < 4; ++r4) {
      #pragma unroll
      for (int v = 0; v < 8; ++v) {          // D layout: row = v + half*8, col = n
        int rm = r4 * 16 + v + half * 8;
        int p = rows[rm];
        if (p >= 0) Hout[(size_t)p * N + n] = (__bf16)fmaxf(accs[r4][v] + bias, 0.0f);
      }
    }
  }
}

// ---------------- kernel 3: layer1 grouped GEMM (576 -> 1024) ----------------
__global__ __launch_bounds__(256) void l1_k(const float* __restrict__ state,    // [B,S]
                                            const float* __restrict__ emb,      // [G,DG]
                                            const __bf16* __restrict__ W1T,     // [E][H][DIN]
                                            const float* __restrict__ b1,       // [E,H]
                                            const int* __restrict__ cnt,
                                            const int* __restrict__ lists,
                                            __bf16* __restrict__ H1) {          // [NPAIR,H]
  constexpr int K = DIN_, N = H_;
  __shared__ alignas(16) __bf16 At[MT][K];   // 72 KB of the 320 KB WGP LDS
  __shared__ int rows[MT];
  int e = blockIdx.y, gi = blockIdx.x;
  int c = cnt[e];
  if (gi * MT >= c) return;
  int t = threadIdx.x;
  if (t < MT) {
    int idx = gi * MT + t;
    rows[t] = (idx < c) ? lists[e * NPAIR + idx] : -1;
  }
  __syncthreads();
  {  // stage A = concat(emb[g], state[b]) as bf16: 4 threads/row, 144 elems each
    int r = t >> 2, q = t & 3;
    int p = rows[r];
    if (p < 0) {
      for (int j = 0; j < 144; ++j) At[r][q * 144 + j] = (__bf16)0.0f;
    } else {
      int b = p >> 7, g = p & (G_ - 1);
      for (int j = 0; j < 144; ++j) {
        int k = q * 144 + j;
        float v = (k < DG_) ? emb[g * DG_ + k] : state[(size_t)b * S_ + (k - DG_)];
        At[r][k] = (__bf16)v;
      }
    }
  }
  __syncthreads();
  gemm_tile_compute<K>(W1T + (size_t)e * N * K, b1 + e * N, At, rows, H1, N, t);
}

// ---------------- kernel 4: layer2 grouped GEMM (1024 -> 1024) ----------------
__global__ __launch_bounds__(256) void l2_k(const __bf16* __restrict__ H1,      // [NPAIR,H]
                                            const __bf16* __restrict__ W2T,     // [E][H][H]
                                            const float* __restrict__ b2,       // [E,H]
                                            const int* __restrict__ cnt,
                                            const int* __restrict__ lists,
                                            __bf16* __restrict__ H2) {          // [NPAIR,H]
  constexpr int K = H_, N = H_;
  __shared__ alignas(16) __bf16 At[MT][K];   // 128 KB of the 320 KB WGP LDS
  __shared__ int rows[MT];
  int e = blockIdx.y, gi = blockIdx.x;
  int c = cnt[e];
  if (gi * MT >= c) return;
  int t = threadIdx.x;
  if (t < MT) {
    int idx = gi * MT + t;
    rows[t] = (idx < c) ? lists[e * NPAIR + idx] : -1;
  }
  __syncthreads();
  {  // gather 64 rows of H1 (2 KB each): 4 threads/row, 32 uint4 (512B) each
    int r = t >> 2, q = t & 3;
    int p = rows[r];
    uint4* dst = (uint4*)&At[r][0];
    if (p < 0) {
      uint4 z = {0u, 0u, 0u, 0u};
      for (int j = 0; j < 32; ++j) dst[q * 32 + j] = z;
    } else {
      const uint4* src = (const uint4*)(H1 + (size_t)p * K);
      for (int j = 0; j < 32; ++j) dst[q * 32 + j] = src[q * 32 + j];
    }
  }
  __syncthreads();
  gemm_tile_compute<K>(W2T + (size_t)e * N * K, b2 + e * N, At, rows, H2, N, t);
}

// ---------------- kernel 5: layer3 (1024 -> 16), fp32 scatter to out ----------------
__global__ __launch_bounds__(32) void l3_k(const __bf16* __restrict__ H2,       // [NPAIR,H]
                                           const __bf16* __restrict__ W3T,      // [E][A][H]
                                           const float* __restrict__ b3,        // [E,A]
                                           const int* __restrict__ cnt,
                                           const int* __restrict__ lists,
                                           float* __restrict__ out) {           // [B,G,A]
  constexpr int K = H_;
  int e = blockIdx.y, gi = blockIdx.x;
  int c = cnt[e];
  if (gi * 16 >= c) return;
  __shared__ int rows[16];
  int lane = threadIdx.x;
  if (lane < 16) {
    int idx = gi * 16 + lane;
    rows[lane] = (idx < c) ? lists[e * NPAIR + idx] : -1;
  }
  __syncthreads();
  int m = lane & 15, half = lane >> 4;
  int pm = rows[m];
  const __bf16* arow = H2 + (size_t)(pm < 0 ? 0 : pm) * K;   // padded rows read row 0, never stored
  const __bf16* wb   = W3T + ((size_t)e * A_ + m) * K + half * 16;
  v8f acc = {0.f, 0.f, 0.f, 0.f, 0.f, 0.f, 0.f, 0.f};
  for (int kb = 0; kb < K; kb += 32) {
    v16bf av = load_a_frag(arow, kb, half);
    v16bf bv = *(const v16bf*)(wb + kb);
    acc = wmma_bf16(av, bv, acc);
  }
  float bias = b3[e * A_ + m];
  #pragma unroll
  for (int v = 0; v < 8; ++v) {
    int rm = v + half * 8;
    int p = rows[rm];
    if (p >= 0) out[(size_t)p * A_ + m] = acc[v] + bias;
  }
}

// ---------------- host launch ----------------
extern "C" void kernel_launch(void* const* d_in, const int* in_sizes, int n_in,
                              void* d_out, int out_size, void* d_ws, size_t ws_size,
                              hipStream_t stream) {
  (void)in_sizes; (void)n_in; (void)out_size; (void)ws_size;
  const float* state  = (const float*)d_in[0];   // [B,S]
  const float* gumbel = (const float*)d_in[1];   // [B,G,E]
  const float* logits = (const float*)d_in[2];   // [G,E]
  const float* emb    = (const float*)d_in[3];   // [G,DG]
  const float* W1     = (const float*)d_in[4];   // [E,DIN,H]
  const float* b1     = (const float*)d_in[5];   // [E,H]
  const float* W2     = (const float*)d_in[6];   // [E,H,H]
  const float* b2     = (const float*)d_in[7];   // [E,H]
  const float* W3     = (const float*)d_in[8];   // [E,H,A]
  const float* b3     = (const float*)d_in[9];   // [E,A]
  float* out = (float*)d_out;

  char* ws = (char*)d_ws;
  int*    cnt   = (int*)(ws + OFF_CNT);
  int*    lists = (int*)(ws + OFF_LIST);
  __bf16* W1T   = (__bf16*)(ws + OFF_W1T);
  __bf16* W2T   = (__bf16*)(ws + OFF_W2T);
  __bf16* W3T   = (__bf16*)(ws + OFF_W3T);
  __bf16* H1    = (__bf16*)(ws + OFF_H1);
  __bf16* H2    = (__bf16*)(ws + OFF_H2);

  init_k<<<1, 32, 0, stream>>>(cnt);
  route_k<<<NPAIR / 256, 256, 0, stream>>>(logits, gumbel, cnt, lists);

  transpose_bf16_k<<<dim3(H_ / 16, DIN_ / 16, E_), 256, 0, stream>>>(W1, W1T, DIN_, H_);
  transpose_bf16_k<<<dim3(H_ / 16, H_   / 16, E_), 256, 0, stream>>>(W2, W2T, H_,   H_);
  transpose_bf16_k<<<dim3(A_ / 16, H_   / 16, E_), 256, 0, stream>>>(W3, W3T, H_,   A_);

  dim3 ggrid(NPAIR / MT, E_, 1);   // worst-case group count per expert; blocks early-exit
  l1_k<<<ggrid, 256, 0, stream>>>(state, emb, W1T, b1, cnt, lists, H1);
  l2_k<<<ggrid, 256, 0, stream>>>(H1, W2T, b2, cnt, lists, H2);
  l3_k<<<dim3(NPAIR / 16, E_, 1), 32, 0, stream>>>(H2, W3T, b3, cnt, lists, out);
}
